// LaminarV1V2Network_60318520705222
// MI455X (gfx1250) — matrix operand order repaired and checked by reading, not compiled, not run on hardware
//
#include <hip/hip_runtime.h>
#include <math.h>

// ---------------- problem constants ----------------
#define NB      1024      // batch
#define TT      512       // timesteps
#define NN      36        // ring size
#define HH      128       // GRU hidden
#define DIN     74        // 2*N + 2
#define KIN     202       // DIN + H
#define KPAD    224       // 7 * 32 (WMMA K tiles)
#define NKT     7
#define BT      16        // batch tile per workgroup (WMMA M)
#define QCOLS   48        // padded head columns: 36 q + 1 pi + 11 zero
#define QOFF    (3 * HH * KPAD)   // u16 offset of head weights in workspace

#define DT_TAU   0.1f
#define DT_TAU_A 0.02f
#define BETA_A   0.5f
#define W_PV     1.0f
#define W_SOM    0.8f
#define FB_GAIN  0.5f

typedef __bf16  v16bf __attribute__((ext_vector_type(16)));
typedef float   v8f   __attribute__((ext_vector_type(8)));

union BPack { unsigned int u[8]; v16bf v; };

__device__ __forceinline__ unsigned short f32_to_bf16(float f) {
    unsigned int u = __float_as_uint(f);
    unsigned int r = (u + 0x7FFFu + ((u >> 16) & 1u)) >> 16;
    return (unsigned short)r;
}
__device__ __forceinline__ float sigmoidf(float x) {
    return 1.0f / (1.0f + __expf(-x));
}

// ---------------------------------------------------------------------------
// Pre-kernel: pack all WMMA weights into workspace as bf16.
//  region 0: GRU gates, [gate][j=0..127][k=0..223], k-major pairs contiguous
//  region 1: head [Wq | Wp | 0], [j=0..47][k=0..127]
// ---------------------------------------------------------------------------
__global__ void convert_weights(const float* __restrict__ Wz,
                                const float* __restrict__ Wr,
                                const float* __restrict__ Wh,
                                const float* __restrict__ Wq,   // (128,36)
                                const float* __restrict__ Wp,   // (128,1)
                                unsigned short* __restrict__ wsW) {
    int idx = blockIdx.x * blockDim.x + threadIdx.x;
    const int gruTotal = 3 * HH * KPAD;
    const int total    = gruTotal + QCOLS * HH;
    if (idx >= total) return;
    if (idx < gruTotal) {
        int g   = idx / (HH * KPAD);
        int rem = idx % (HH * KPAD);
        int j   = rem / KPAD;
        int k   = rem % KPAD;
        const float* W = (g == 0) ? Wz : (g == 1) ? Wr : Wh;
        float v = (k < KIN) ? W[k * HH + j] : 0.0f;
        wsW[idx] = f32_to_bf16(v);
    } else {
        int rem = idx - gruTotal;
        int j   = rem / HH;           // head output column (0..47)
        int k   = rem % HH;           // hidden index
        float v = (j < NN) ? Wq[k * NN + j] : (j == NN) ? Wp[k] : 0.0f;
        wsW[idx] = f32_to_bf16(v);
    }
}

// ---------------------------------------------------------------------------
// Main recurrent kernel: one workgroup = 16 batch rows, 8 wave32s.
// Wave w owns GRU columns [16w, 16w+16); waves 0-2 additionally own the
// head N-tiles. GRU + head matmuls via v_wmma_f32_16x16x32_bf16.
// ---------------------------------------------------------------------------
__global__ void __launch_bounds__(256, 1)
laminar_step_kernel(const float* __restrict__ packed,   // (B,T,74)
                    const float* __restrict__ W_l4,     // (36,36)
                    const float* __restrict__ W_l23,
                    const float* __restrict__ W_som,
                    const float* __restrict__ bz,
                    const float* __restrict__ br,
                    const float* __restrict__ bh,
                    const float* __restrict__ bq,       // (36)
                    const float* __restrict__ bp,       // (1)
                    const unsigned short* __restrict__ wsW,
                    float* __restrict__ out)            // (B,T,36)
{
    __shared__ __align__(16) unsigned short s_x[BT][KPAD];  // bf16 xh / xrh
    __shared__ __align__(16) unsigned short s_hb[BT][HH];   // bf16 h (K-aligned copy)
    __shared__ float s_in[BT][DIN];
    __shared__ float s_l4[BT][NN],  s_l23[BT][NN],  s_som[BT][NN],  s_ad[BT][NN];
    __shared__ float s_n4[BT][NN],  s_n23[BT][NN],  s_nsom[BT][NN], s_nad[BT][NN];
    __shared__ float s_pv[BT], s_npv[BT], s_pi[BT], s_pilog[BT];
    __shared__ float s_q[BT][NN];
    __shared__ float s_Wl4[NN][NN], s_Wl23[NN][NN], s_Wsom[NN][NN];
    __shared__ float s_bz[HH], s_br[HH], s_bh[HH], s_bq[NN];

    const int tid  = threadIdx.x;
    const int wave = tid >> 5;
    const int lane = tid & 31;
    const int col  = lane & 15;               // B-operand column within tile
    const int ncol = wave * 16 + col;         // global GRU column for this lane
    const int hiHalf = (lane >> 4);           // 0: lanes 0-15, 1: lanes 16-31
    const int b0 = blockIdx.x * BT;

    // ---- init LDS state ----
    for (int i = tid; i < BT * KPAD; i += 256) s_x[i / KPAD][i % KPAD] = 0;
    for (int i = tid; i < BT * HH; i += 256) s_hb[i / HH][i % HH] = 0;
    for (int i = tid; i < BT * NN; i += 256) {
        int b = i / NN, j = i % NN;
        s_l4[b][j] = 0.f; s_l23[b][j] = 0.f; s_som[b][j] = 0.f; s_ad[b][j] = 0.f;
    }
    if (tid < BT) { s_pv[tid] = 0.f; s_npv[tid] = 0.f; s_pi[tid] = 0.f; s_pilog[tid] = 0.f; }
    for (int i = tid; i < NN * NN; i += 256) {
        s_Wl4 [i / NN][i % NN] = W_l4 [i];
        s_Wl23[i / NN][i % NN] = W_l23[i];
        s_Wsom[i / NN][i % NN] = W_som[i];
    }
    for (int i = tid; i < HH; i += 256) { s_bz[i] = bz[i]; s_br[i] = br[i]; s_bh[i] = bh[i]; }
    if (tid < NN) s_bq[tid] = bq[tid];

    // ---- preload persistent B-operand weight registers (bf16) ----
    // B layout (32x16 K x N): lane (0-15)=col N, K pairs v<4: 2v, v>=4: 16+2(v-4); lanes 16-31 K+=8.
    v16bf Bz[NKT], Br[NKT], Bh[NKT], Bq[4];
    {
        const unsigned int* w32 = (const unsigned int*)wsW;
        const int j = ncol;
        const int khalf = hiHalf * 8;
#pragma unroll
        for (int kt = 0; kt < NKT; ++kt) {
            BPack pz, pr, ph;
#pragma unroll
            for (int v = 0; v < 8; ++v) {
                int k = kt * 32 + khalf + ((v < 4) ? 2 * v : 16 + 2 * (v - 4));
                int base = j * KPAD + k;
                pz.u[v] = w32[(0 * HH * KPAD + base) >> 1];
                pr.u[v] = w32[(1 * HH * KPAD + base) >> 1];
                ph.u[v] = w32[(2 * HH * KPAD + base) >> 1];
            }
            Bz[kt] = pz.v; Br[kt] = pr.v; Bh[kt] = ph.v;
        }
        // head B regs: valid for waves 0-2; other waves load tile 0 (unused)
        const int jq = ((wave < 3) ? wave : 0) * 16 + col;
#pragma unroll
        for (int kt = 0; kt < 4; ++kt) {
            BPack pq;
#pragma unroll
            for (int v = 0; v < 8; ++v) {
                int k = kt * 32 + khalf + ((v < 4) ? 2 * v : 16 + 2 * (v - 4));
                pq.u[v] = w32[(QOFF + jq * HH + k) >> 1];
            }
            Bq[kt] = pq.v;
        }
    }

    // h state in registers (C-matrix layout): element i of this lane is
    // h[m][ncol] with m = i + 8*hiHalf.
    float hreg[8];
#pragma unroll
    for (int i = 0; i < 8; ++i) hreg[i] = 0.f;

    const float bp0 = bp[0];
    __syncthreads();

    for (int t = 0; t < TT; ++t) {
        // ---- phase 1: stage inputs for this step ----
        for (int i = tid; i < BT * DIN; i += 256) {
            int b = i / DIN, k = i % DIN;
            s_in[b][k] = packed[((size_t)(b0 + b) * TT + t) * DIN + k];
        }
        if (t + 1 < TT) {   // prefetch next step's rows (global_prefetch_b8)
            int m = tid >> 4, off = (tid & 15) * 5;
            if (m < BT && off < DIN)
                __builtin_prefetch(&packed[((size_t)(b0 + m) * TT + (t + 1)) * DIN + off], 0, 1);
        }
        __syncthreads();

        // ---- phase 2: L4 dynamics + adaptation; build x-part of xh (bf16) ----
        for (int idx = tid; idx < BT * NN; idx += 256) {
            int b = idx / NN, i = idx % NN;
            float dot = 0.f;
#pragma unroll 4
            for (int j = 0; j < NN; ++j) dot = fmaf(s_l4[b][j], s_Wl4[i][j], dot);
            float drive = s_in[b][i] + dot - W_PV * s_pv[b] - s_ad[b][i];
            float r4    = s_l4[b][i] + DT_TAU * (-s_l4[b][i] + fmaxf(drive, 0.f));
            s_n4[b][i]  = r4;
            s_nad[b][i] = s_ad[b][i] + DT_TAU_A * (-s_ad[b][i] + BETA_A * r4);
        }
        for (int idx = tid; idx < BT * DIN; idx += 256) {
            int b = idx / DIN, k = idx % DIN;
            float v = (k < NN) ? s_l23[b][k] : s_in[b][k];   // [r_l23, cue, task]
            s_x[b][k] = f32_to_bf16(v);
        }
        __syncthreads();

        // ---- phase 3: PV (scalar per row) while waves run z/r WMMAs ----
        if (tid < BT) {
            float m4 = 0.f, m23 = 0.f;
            for (int j = 0; j < NN; ++j) { m4 += s_n4[tid][j]; m23 += s_l23[tid][j]; }
            float pvin = (m4 + m23) * (1.0f / NN);
            s_npv[tid] = s_pv[tid] + DT_TAU * (-s_pv[tid] + fmaxf(pvin, 0.f));
        }

        // z and r gates: 16x16 tile per wave, K = 224 (7 WMMAs each)
        v8f zacc = {}; v8f racc = {};
        {
            const unsigned int* px = (const unsigned int*)&s_x[lane & 15][0];
            const int koff = hiHalf ? 8 : 0;
#pragma unroll
            for (int kt = 0; kt < NKT; ++kt) {
                BPack a;
#pragma unroll
                for (int v = 0; v < 8; ++v) {
                    int kk = kt * 32 + koff + ((v < 4) ? 2 * v : 16 + 2 * (v - 4));
                    a.u[v] = px[kk >> 1];
                }
                zacc = __builtin_amdgcn_wmma_f32_16x16x32_bf16(
                           false, a.v, false, Bz[kt], (short)0, zacc, false, false);
                racc = __builtin_amdgcn_wmma_f32_16x16x32_bf16(
                           false, a.v, false, Br[kt], (short)0, racc, false, false);
            }
        }
        float zg[8], rg[8];
#pragma unroll
        for (int i = 0; i < 8; ++i) {
            zg[i] = sigmoidf(zacc[i] + s_bz[ncol]);
            rg[i] = sigmoidf(racc[i] + s_br[ncol]);
        }
        __syncthreads();    // everyone done reading xh

        // write r*h into h-slot of s_x (now xrh)
#pragma unroll
        for (int i = 0; i < 8; ++i) {
            int m = i + 8 * hiHalf;
            s_x[m][DIN + ncol] = f32_to_bf16(rg[i] * hreg[i]);
        }
        __syncthreads();

        // hh candidate
        v8f hacc = {};
        {
            const unsigned int* px = (const unsigned int*)&s_x[lane & 15][0];
            const int koff = hiHalf ? 8 : 0;
#pragma unroll
            for (int kt = 0; kt < NKT; ++kt) {
                BPack a;
#pragma unroll
                for (int v = 0; v < 8; ++v) {
                    int kk = kt * 32 + koff + ((v < 4) ? 2 * v : 16 + 2 * (v - 4));
                    a.u[v] = px[kk >> 1];
                }
                hacc = __builtin_amdgcn_wmma_f32_16x16x32_bf16(
                           false, a.v, false, Bh[kt], (short)0, hacc, false, false);
            }
        }
#pragma unroll
        for (int i = 0; i < 8; ++i) {
            float hh = tanhf(hacc[i] + s_bh[ncol]);
            hreg[i]  = (1.f - zg[i]) * hreg[i] + zg[i] * hh;
        }
        __syncthreads();    // everyone done reading xrh

        // publish new h (bf16): into xh slot for next step and aligned head copy
#pragma unroll
        for (int i = 0; i < 8; ++i) {
            int m = i + 8 * hiHalf;
            unsigned short hb = f32_to_bf16(hreg[i]);
            s_x[m][DIN + ncol] = hb;
            s_hb[m][ncol]      = hb;
        }
        __syncthreads();

        // ---- phase 4: heads via WMMA (waves 0-2: q tiles 0-2; col 36 = pi) ----
        if (wave < 3) {             // wave-uniform branch: EXEC stays all-ones
            v8f qacc = {};
            const unsigned int* ph = (const unsigned int*)&s_hb[lane & 15][0];
            const int koff = hiHalf ? 8 : 0;
#pragma unroll
            for (int kt = 0; kt < 4; ++kt) {
                BPack a;
#pragma unroll
                for (int v = 0; v < 8; ++v) {
                    int kk = kt * 32 + koff + ((v < 4) ? 2 * v : 16 + 2 * (v - 4));
                    a.u[v] = ph[kk >> 1];
                }
                qacc = __builtin_amdgcn_wmma_f32_16x16x32_bf16(
                           false, a.v, false, Bq[kt], (short)0, qacc, false, false);
            }
            const int gcol = wave * 16 + col;
#pragma unroll
            for (int i = 0; i < 8; ++i) {
                int m = i + 8 * hiHalf;
                if (gcol < NN)       s_q[m][gcol] = qacc[i] + s_bq[gcol];
                else if (gcol == NN) s_pilog[m]   = qacc[i] + bp0;
            }
        }
        __syncthreads();

        // softmax over 36 per row + pi sigmoid
        if (tid < BT) {
            float mx = s_q[tid][0];
            for (int i = 1; i < NN; ++i) mx = fmaxf(mx, s_q[tid][i]);
            float sum = 0.f;
            for (int i = 0; i < NN; ++i) { float e = __expf(s_q[tid][i] - mx); s_q[tid][i] = e; sum += e; }
            float inv = 1.0f / sum;
            for (int i = 0; i < NN; ++i) s_q[tid][i] *= inv;
            s_pi[tid] = sigmoidf(s_pilog[tid]);
        }
        __syncthreads();

        // ---- phase 5: SOM + L23 dynamics, emit output ----
        for (int idx = tid; idx < BT * NN; idx += 256) {
            int b = idx / NN, i = idx % NN;
            float qp = s_q[b][i] * s_pi[b];                       // q_pred * pi_eff
            float sdot = 0.f, ldot = 0.f;
#pragma unroll 4
            for (int j = 0; j < NN; ++j) {
                sdot = fmaf(s_som[b][j], s_Wsom[i][j], sdot);
                ldot = fmaf(s_l23[b][j], s_Wl23[i][j], ldot);
            }
            float somn = s_som[b][i] + DT_TAU * (-s_som[b][i] + fmaxf(qp + sdot, 0.f));
            s_nsom[b][i] = somn;
            float l4to  = s_n4[b][i] - qp;                        // ALPHA = 1
            float tmpl  = 1.0f + FB_GAIN * qp;
            float drive = l4to * tmpl + ldot - W_SOM * somn - W_PV * s_npv[b];
            float l23n  = s_l23[b][i] + DT_TAU * (-s_l23[b][i] + fmaxf(drive, 0.f));
            s_n23[b][i] = l23n;
            out[((size_t)(b0 + b) * TT + t) * NN + i] = l23n;
        }
        __syncthreads();

        // ---- commit state ----
        for (int idx = tid; idx < BT * NN; idx += 256) {
            int b = idx / NN, i = idx % NN;
            s_l4[b][i]  = s_n4[b][i];
            s_ad[b][i]  = s_nad[b][i];
            s_som[b][i] = s_nsom[b][i];
            s_l23[b][i] = s_n23[b][i];
        }
        if (tid < BT) s_pv[tid] = s_npv[tid];
        __syncthreads();
    }
}

// ---------------------------------------------------------------------------
extern "C" void kernel_launch(void* const* d_in, const int* in_sizes, int n_in,
                              void* d_out, int out_size, void* d_ws, size_t ws_size,
                              hipStream_t stream) {
    const float* packed = (const float*)d_in[0];
    const float* W_l4   = (const float*)d_in[1];
    const float* W_l23  = (const float*)d_in[2];
    const float* W_som  = (const float*)d_in[3];
    const float* Wz     = (const float*)d_in[4];
    const float* Wr     = (const float*)d_in[5];
    const float* Wh     = (const float*)d_in[6];
    const float* bzp    = (const float*)d_in[7];
    const float* brp    = (const float*)d_in[8];
    const float* bhp    = (const float*)d_in[9];
    const float* Wq     = (const float*)d_in[10];
    const float* bq     = (const float*)d_in[11];
    const float* Wp     = (const float*)d_in[12];
    const float* bp     = (const float*)d_in[13];
    // d_in[14]=Ws, d_in[15]=bs: state_logits are dead code in the reference.

    unsigned short* wsW = (unsigned short*)d_ws;   // bf16 weights

    const int convTotal = 3 * HH * KPAD + QCOLS * HH;
    convert_weights<<<(convTotal + 255) / 256, 256, 0, stream>>>(Wz, Wr, Wh, Wq, Wp, wsW);

    laminar_step_kernel<<<NB / BT, 256, 0, stream>>>(
        packed, W_l4, W_l23, W_som, bzp, brp, bhp, bq, bp, wsW,
        (float*)d_out);
}
